// InfoCNECauchy_38577396253108
// MI455X (gfx1250) — matrix operand order repaired and checked by reading, not compiled, not run on hardware
//
#include <hip/hip_runtime.h>
#include <hip/hip_bf16.h>

// ---------------------------------------------------------------------------
// InfoCNE-Cauchy loss, N2=8192 rows, D=512, fp32 in, scalar fp32 out.
//
//   G = F F^T  (bf16 WMMA, f32 accum)          -- compute bound, 68.7 GFLOP
//   d2 = max(sq_i + sq_j - 2 G_ij, 0)
//   sim = 1 / (d2 / t^2 + 1)
//   out = mean_i log(sum_{j!=i} sim_ij) - mean_i log(sim_{i,partner(i)})
//
// partner(i) = i ^ 4096 (two views of 4096 samples).
// ---------------------------------------------------------------------------

typedef __attribute__((ext_vector_type(16))) __bf16 v16bf;
typedef __attribute__((ext_vector_type(8)))  float  v8f;
typedef __attribute__((ext_vector_type(4)))  unsigned int v4u;
typedef __attribute__((ext_vector_type(4)))  float  v4f;

#define N2      8192
#define DIM     512
#define NSPLIT  8                      // column splits (deterministic partial sums)
#define COLS_PER_SPLIT (N2 / NSPLIT)   // 1024
#define NTILES  (COLS_PER_SPLIT / 16)  // 64 column tiles per wave
#define KTILES  16                     // 512 / 32
#define PIPE    4                      // B-fragment pipeline depth (8 loads in flight)
#define INV_T2  (1.0f / (0.07f * 0.07f))

// ---------------------------------------------------------------------------
// Kernel 1: fp32 -> bf16 copy + exact fp32 row norms. One wave per row.
// ---------------------------------------------------------------------------
__global__ __launch_bounds__(256) void cauchy_prep(const float* __restrict__ F,
                                                   __bf16* __restrict__ Fbf,
                                                   float* __restrict__ sq) {
    const int wave = threadIdx.x >> 5;
    const int lane = threadIdx.x & 31;
    const int row  = blockIdx.x * 8 + wave;       // 1024 blocks * 8 waves = 8192
    const float*  rp = F   + (size_t)row * DIM + lane * 16;
    __bf16*       op = Fbf + (size_t)row * DIM + lane * 16;

    union { __bf16 b[16]; v4u u[2]; } t;
    float ss = 0.0f;
#pragma unroll
    for (int c = 0; c < 16; c += 4) {
        v4f f = *(const v4f*)(rp + c);
        ss += f.x * f.x + f.y * f.y + f.z * f.z + f.w * f.w;
        t.b[c + 0] = (__bf16)f.x;
        t.b[c + 1] = (__bf16)f.y;
        t.b[c + 2] = (__bf16)f.z;
        t.b[c + 3] = (__bf16)f.w;
    }
    *(v4u*)(op + 0) = t.u[0];
    *(v4u*)(op + 8) = t.u[1];

#pragma unroll
    for (int m = 16; m >= 1; m >>= 1) ss += __shfl_xor(ss, m, 32);
    if (lane == 0) sq[row] = ss;
}

// ---------------------------------------------------------------------------
// Kernel 2: WMMA GEMM + Cauchy epilogue, software-pipelined B stream.
//   block = 256 threads = 8 waves; wave w owns rows [i0, i0+16).
//   grid = (64 slab groups, NSPLIT column ranges).
//   A panel (16x512 bf16) resident in 128 VGPRs; B fragments flow through a
//   4-deep circular register buffer (8 global_load_b128 in flight behind the
//   v_wmma pipe). Column norms live in LDS so the epilogue never touches
//   LOADcnt -> no full-drain s_wait_loadcnt 0x0 inside the steady loop.
// ---------------------------------------------------------------------------
__device__ __forceinline__ v16bf load16bf(const __bf16* p) {
    union { v4u u[2]; v16bf v; } r;
    r.u[0] = *(const v4u*)(p);
    r.u[1] = *(const v4u*)(p + 16);
    return r.v;
}

__global__ __launch_bounds__(256, 1) void cauchy_gemm(const __bf16* __restrict__ F,
                                                      const float*  __restrict__ sq,
                                                      float* __restrict__ rowPartial,
                                                      float* __restrict__ simPartner) {
    __shared__ float sqCol[COLS_PER_SPLIT];   // 4 KB: this split's column norms

    const int lane = threadIdx.x & 31;
    const int wave = threadIdx.x >> 5;
    const int hi   = lane >> 4;          // half-wave select (ISA fragment layout)
    const int lr   = lane & 15;
    const int i0   = (blockIdx.x * 8 + wave) << 4;      // row slab base
    const int split = blockIdx.y;
    const int jBeg = split * COLS_PER_SPLIT;

    // stage column norms in LDS (read later via DScnt, not LOADcnt)
    for (int c = threadIdx.x; c < COLS_PER_SPLIT; c += 256)
        sqCol[c] = sq[jBeg + c];
    __syncthreads();

    // --- resident A fragments: 16 k-tiles of 16x32 bf16 ---
    // lanes 0-15 : row M=lr, K = {kb..kb+7, kb+16..kb+23}
    // lanes 16-31: row M=lr, K = {kb+8..kb+15, kb+24..kb+31}
    const __bf16* aRowP = F + (size_t)(i0 + lr) * DIM;
    v16bf aFrag[KTILES];
#pragma unroll
    for (int kt = 0; kt < KTILES; ++kt) {
        const int c0 = kt * 32 + (hi ? 8 : 0);
        union { v4u u[2]; v16bf v; } r;
        r.u[0] = *(const v4u*)(aRowP + c0);
        r.u[1] = *(const v4u*)(aRowP + c0 + 16);
        aFrag[kt] = r.v;
    }

    // row norms for the 8 rows this half-wave produces (C layout: M = v + 8*hi)
    float sqM[8];
#pragma unroll
    for (int v = 0; v < 8; ++v) sqM[v] = sq[i0 + v + hi * 8];

    float rowpart[8] = {0.f, 0.f, 0.f, 0.f, 0.f, 0.f, 0.f, 0.f};

    // B fragment address for flattened step t (t = tile*16 + kt):
    //   row (jBeg + tile*16 + lr) of F, K = {8hi..8hi+7, 16+8hi..16+8hi+7}.
    const int bKoff = (hi ? 8 : 0);
    auto bAddr = [&](int t) -> const __bf16* {
        const int jN = jBeg + ((t >> 4) << 4);
        const int kN = (t & 15) * 32;
        return F + (size_t)(jN + lr) * DIM + bKoff + kN;
    };

    const int numT = NTILES * KTILES;    // 1024 flattened (tile, ktile) steps

    // prime the pipeline: PIPE fragments in flight
    v16bf bbuf[PIPE];
#pragma unroll
    for (int p = 0; p < PIPE; ++p) bbuf[p] = load16bf(bAddr(p));

    for (int tile = 0; tile < NTILES; ++tile) {
        const int j0 = jBeg + tile * 16;
        // L2 -> near-cache prefetch two tiles ahead (global_prefetch_b8)
        if (tile + 2 < NTILES)
            __builtin_prefetch(F + (size_t)(j0 + 32 + lr) * DIM, 0, 3);

        v8f acc = {0.f, 0.f, 0.f, 0.f, 0.f, 0.f, 0.f, 0.f};
#pragma unroll
        for (int kt = 0; kt < KTILES; ++kt) {
            v16bf b = bbuf[kt & (PIPE - 1)];
            const int tNext = tile * KTILES + kt + PIPE;
            if (tNext < numT)            // uniform; false only on last tile
                bbuf[kt & (PIPE - 1)] = load16bf(bAddr(tNext));
            acc = __builtin_amdgcn_wmma_f32_16x16x32_bf16(
                false, aFrag[kt], false, b, (short)0, acc, false, false);
        }

        // epilogue: Cauchy kernel, exclude diagonal, capture partner sim.
        // sqN comes from LDS -> ds_load_b32 (DScnt), keeps B pipeline intact.
        const int   gN  = j0 + lr;
        const float sqN = sqCol[tile * 16 + lr];
#pragma unroll
        for (int v = 0; v < 8; ++v) {
            const int gM = i0 + v + hi * 8;
            float d2  = fmaxf(sqM[v] + sqN - 2.0f * acc[v], 0.0f);
            float sim = 1.0f / (d2 * INV_T2 + 1.0f);
            rowpart[v] += (gN == gM) ? 0.0f : sim;
            if (gN == (gM ^ 4096)) simPartner[gM] = sim;   // unique writer
        }
    }

    // reduce row sums across the 16 lanes of each half-wave, store partial
#pragma unroll
    for (int v = 0; v < 8; ++v) {
        float s = rowpart[v];
#pragma unroll
        for (int m = 8; m >= 1; m >>= 1) s += __shfl_xor(s, m, 16);
        if (lr == 0)
            rowPartial[(size_t)split * N2 + i0 + v + hi * 8] = s;
    }
}

// ---------------------------------------------------------------------------
// Kernel 3: deterministic final reduction -> scalar loss.
// ---------------------------------------------------------------------------
__global__ __launch_bounds__(256) void cauchy_finish(const float* __restrict__ rowPartial,
                                                     const float* __restrict__ simPartner,
                                                     float* __restrict__ out) {
    __shared__ float red[256];
    const int tid = threadIdx.x;
    float local = 0.0f;
    for (int r = tid; r < N2; r += 256) {
        float S = 0.0f;
#pragma unroll
        for (int s = 0; s < NSPLIT; ++s) S += rowPartial[(size_t)s * N2 + r];
        local += logf(S) - logf(simPartner[r]);
    }
    red[tid] = local;
    __syncthreads();
    for (int m = 128; m >= 1; m >>= 1) {
        if (tid < m) red[tid] += red[tid + m];
        __syncthreads();
    }
    if (tid == 0) out[0] = red[0] / (float)N2;   // = neg - pos = -(pos - neg)
}

// ---------------------------------------------------------------------------
extern "C" void kernel_launch(void* const* d_in, const int* in_sizes, int n_in,
                              void* d_out, int out_size, void* d_ws, size_t ws_size,
                              hipStream_t stream) {
    (void)in_sizes; (void)n_in; (void)out_size; (void)ws_size;
    const float* feats = (const float*)d_in[0];   // (8192, 512) fp32; labels unused

    char* ws = (char*)d_ws;
    __bf16* Fbf       = (__bf16*)ws;                                   // 8 MB
    float*  sq        = (float*)(ws + (size_t)N2 * DIM * 2);           // 32 KB
    float*  rowPart   = sq + N2;                                       // 256 KB
    float*  simPart   = rowPart + (size_t)NSPLIT * N2;                 // 32 KB
    float*  out       = (float*)d_out;

    cauchy_prep<<<N2 / 8, 256, 0, stream>>>(feats, Fbf, sq);

    dim3 grid(N2 / 16 / 8, NSPLIT);   // (64, 8) blocks, 8 waves each
    cauchy_gemm<<<grid, 256, 0, stream>>>(Fbf, sq, rowPart, simPart);

    cauchy_finish<<<1, 256, 0, stream>>>(rowPart, simPart, out);
}
